// dspn_3x3_naive_40338332844078
// MI455X (gfx1250) — matrix operand order repaired and checked by reading, not compiled, not run on hardware
//
#include <hip/hip_runtime.h>

typedef _Float16 f16;
typedef __attribute__((ext_vector_type(16))) _Float16 v16h;
typedef __attribute__((ext_vector_type(8)))  float    v8f;

#define B_   2
#define H_   352
#define W_   1216
#define NPIX (B_ * H_ * W_)       // 856064
#define TW   64                   // pixels per workgroup (4 waves x 16)
#define NTX  (W_ / TW)            // 19 (exact)
#define ROWS 4                    // rows per workgroup (H_/ROWS = 88 exact)
#define GTW  (TW + 2)             // 66, halo-tile row width
#define GTR  (ROWS + 2)           // 6, halo-tile rows per channel
#define GT_ZPAD (9 * GTR * GTW)   // 3564: base of zero-pad region
#define GT_TOT  (GT_ZPAD + (ROWS - 1) * GTW + TW + 2)  // 3828 halves

// Compile-time map: im2col K index -> LDS offset within guidance tile
// (relative to pixel p and row ry*GTW). Invalid K (>=81) -> zero-pad region.
constexpr int bofs(int k) {
    return (k < 81)
        ? ((k / 9) * GTR + (k % 9) / 3) * GTW + ((k % 9) % 3)
        : GT_ZPAD;
}

// ---------------------------------------------------------------------------
// Kernel 1: oa = conv3x3(guidance, conv_w) + conv_b  (im2col GEMM on WMMA f16)
//   offset1 = f16(oa[:, :18]),  aff1 = f16(softmax(oa[:, 18:27]))
// One workgroup = 4 waves x 16 pixels x ROWS rows; A fragments register-resident.
// ---------------------------------------------------------------------------
__global__ void __launch_bounds__(128)
conv_offset_aff_wmma(const float* __restrict__ guidance,
                     const float* __restrict__ conv_w,
                     const float* __restrict__ conv_b,
                     f16* __restrict__ off_out,
                     f16* __restrict__ aff_out)
{
    __shared__ f16   Wh[32 * 96];        // padded weights [M=32][K=96]
    __shared__ f16   gt[GT_TOT];         // guidance halo tile + zero pad
    __shared__ float oas[4][32][16];     // per-wave C scratch [wave][ch][pixel]

    const int tid = threadIdx.x;
    const int x0  = blockIdx.x * TW;
    const int y0  = blockIdx.y * ROWS;
    const int b   = blockIdx.z;

    // --- stage padded f16 weights: Wh[o][k] = conv_w[o*81 + k]
    for (int idx = tid; idx < 32 * 96; idx += 128) {
        int o = idx / 96, k = idx % 96;
        float v = (o < 27 && k < 81) ? conv_w[o * 81 + k] : 0.0f;
        Wh[idx] = (f16)v;
    }
    // --- stage guidance halo tile gt[c][r][rx] (+ zero-pad tail region)
    for (int idx = tid; idx < GT_TOT; idx += 128) {
        float v = 0.0f;
        if (idx < GT_ZPAD) {
            int c  = idx / (GTR * GTW);
            int r  = (idx / GTW) % GTR;
            int rx = idx % GTW;
            int gy = y0 - 1 + r;
            int gx = x0 - 1 + rx;
            if (gy >= 0 && gy < H_ && gx >= 0 && gx < W_)
                v = guidance[((b * 9 + c) * H_ + gy) * W_ + gx];
        }
        gt[idx] = (f16)v;
    }
    __syncthreads();

    const int wave = tid >> 5;
    const int lane = tid & 31;
    const int half = lane >> 4;
    const int l16  = lane & 15;
    const int px0  = wave * 16;          // this wave's pixel base within tile

    // --- A fragments (weights), register-resident across all rows.
    // 16-bit A 16x32 layout: lanes 0-15 hold K {0..7,16..23}, lanes 16-31 {8..15,24..31}.
    v16h a0f[3], a1f[3];
    #pragma unroll
    for (int kt = 0; kt < 3; ++kt) {
        const int kbA = kt * 32 + (half ? 8 : 0);
        #pragma unroll
        for (int i = 0; i < 8; ++i) {
            a0f[kt][i]     = Wh[l16 * 96 + kbA + i];
            a0f[kt][8 + i] = Wh[l16 * 96 + kbA + 16 + i];
            a1f[kt][i]     = Wh[(16 + l16) * 96 + kbA + i];
            a1f[kt][8 + i] = Wh[(16 + l16) * 96 + kbA + 16 + i];
        }
    }

    const int p = px0 + l16;             // B column (pixel within tile)

    for (int ry = 0; ry < ROWS; ++ry) {
        const int pr = p + ry * GTW;     // hoisted per-row pixel base

        v8f c0 = {};
        v8f c1 = {};
        #pragma unroll
        for (int kt = 0; kt < 3; ++kt) {
            // B fragment (32x16): lanes 0-15 K {0..7,16..23}, lanes 16-31 K {8..15,24..31}.
            // LDS offset selected between two compile-time constants per element.
            v16h bf;
            #pragma unroll
            for (int j = 0; j < 16; ++j) {
                const int o0 = bofs(kt * 32 + (j < 8 ? j : j + 8));
                const int o1 = bofs(kt * 32 + (j < 8 ? j + 8 : j + 16));
                bf[j] = gt[(half ? o1 : o0) + pr];
            }
            c0 = __builtin_amdgcn_wmma_f32_16x16x32_f16(false, a0f[kt], false, bf, (short)0, c0, false, false);
            c1 = __builtin_amdgcn_wmma_f32_16x16x32_f16(false, a1f[kt], false, bf, (short)0, c1, false, false);
        }

        // C layout: VGPR g -> (lanes 0-15: M=g) (lanes 16-31: M=g+8), N=lane%16
        #pragma unroll
        for (int g = 0; g < 8; ++g) {
            int row = half ? (g + 8) : g;
            oas[wave][row][l16]      = c0[g];
            oas[wave][row + 16][l16] = c1[g];
        }
        // oas is private per wave: intra-wave DS ordering (dscnt) suffices, no barrier.

        const int y = y0 + ry;
        if (lane < 16) {
            // lanes 0-15: softmax over channels 18..26 -> aff1 (f16)
            const int pp = lane;
            const int x  = x0 + wave * 16 + pp;
            float v[9], m = -3.4e38f;
            #pragma unroll
            for (int j = 0; j < 9; ++j) {
                v[j] = oas[wave][18 + j][pp] + conv_b[18 + j];
                m = fmaxf(m, v[j]);
            }
            float s = 0.0f;
            #pragma unroll
            for (int j = 0; j < 9; ++j) { v[j] = expf(v[j] - m); s += v[j]; }
            const float inv = 1.0f / s;
            #pragma unroll
            for (int j = 0; j < 9; ++j)
                aff_out[((b * 9 + j) * H_ + y) * W_ + x] = (f16)(v[j] * inv);
        } else {
            // lanes 16-31: channels 0..17 -> offset1 (f16)
            const int pp = lane - 16;
            const int x  = x0 + wave * 16 + pp;
            #pragma unroll
            for (int ch = 0; ch < 18; ++ch)
                off_out[((b * 18 + ch) * H_ + y) * W_ + x] =
                    (f16)(oas[wave][ch][pp] + conv_b[ch]);
        }
    }
}

// ---------------------------------------------------------------------------
// Kernel 2: one propagation step
//   fr_out = (1-conf) * deform_conv(fr_prev, off, aff, w9, bias) + conf * feat_fix
// Exact 3D grid: no div/mod. 18 launches; working set is L2-resident.
// ---------------------------------------------------------------------------
__global__ void __launch_bounds__(64)
prop_step(const float* __restrict__ fr_prev,
          const float* __restrict__ confidence,
          const float* __restrict__ feat_fix,
          const f16*   __restrict__ off,
          const f16*   __restrict__ aff,
          const float* __restrict__ w9,
          const float* __restrict__ bias,
          float* __restrict__ fr_out,
          float* __restrict__ pred)     // non-null only on the final iteration
{
    const int x = blockIdx.x * 64 + threadIdx.x;
    const int y = blockIdx.y;
    const int b = blockIdx.z;
    const int CH = H_ * W_;
    const int i  = (b * H_ + y) * W_ + x;

    const float cs = 1.0f / (1.0f + expf(-confidence[i]));
    const float ff = feat_fix[i];
    const float sg = (ff > 0.0f) ? 1.0f : ((ff < 0.0f) ? -1.0f : 0.0f);
    const float conf = sg * cs;

    const float* img = fr_prev + b * CH;
    const int obase = (b * 18) * CH + y * W_ + x;
    const int abase = (b * 9)  * CH + y * W_ + x;

    float acc = bias[0];
    #pragma unroll
    for (int k = 0; k < 9; ++k) {
        const float dy = (float)off[obase + (2 * k    ) * CH];
        const float dx = (float)off[obase + (2 * k + 1) * CH];
        const float mk = (float)aff[abase + k * CH];
        const float py = (float)y + (float)(k / 3 - 1) + dy;
        const float px = (float)x + (float)(k % 3 - 1) + dx;
        const float y0f = floorf(py), x0f = floorf(px);
        const float ty = py - y0f, tx = px - x0f;
        const int iy0 = (int)y0f, ix0 = (int)x0f;
        const bool y0ok = (iy0     >= 0) && (iy0     <= H_ - 1);
        const bool y1ok = (iy0 + 1 >= 0) && (iy0 + 1 <= H_ - 1);
        const bool x0ok = (ix0     >= 0) && (ix0     <= W_ - 1);
        const bool x1ok = (ix0 + 1 >= 0) && (ix0 + 1 <= W_ - 1);
        float v00 = 0.0f, v01 = 0.0f, v10 = 0.0f, v11 = 0.0f;
        if (y0ok && x0ok) v00 = img[iy0 * W_ + ix0];
        if (y0ok && x1ok) v01 = img[iy0 * W_ + ix0 + 1];
        if (y1ok && x0ok) v10 = img[(iy0 + 1) * W_ + ix0];
        if (y1ok && x1ok) v11 = img[(iy0 + 1) * W_ + ix0 + 1];
        const float s = (1.0f - ty) * ((1.0f - tx) * v00 + tx * v01)
                      +         ty  * ((1.0f - tx) * v10 + tx * v11);
        acc += s * mk * w9[k];
    }
    const float out = (1.0f - conf) * acc + conf * ff;
    fr_out[i] = out;
    if (pred) pred[i] = out;
}

// ---------------------------------------------------------------------------
extern "C" void kernel_launch(void* const* d_in, const int* in_sizes, int n_in,
                              void* d_out, int out_size, void* d_ws, size_t ws_size,
                              hipStream_t stream)
{
    (void)in_sizes; (void)n_in; (void)out_size; (void)d_ws; (void)ws_size;

    const float* feat_init  = (const float*)d_in[0];
    const float* guidance   = (const float*)d_in[1];
    const float* confidence = (const float*)d_in[2];
    const float* feat_fix   = (const float*)d_in[3];
    const float* conv_w     = (const float*)d_in[4];
    const float* conv_b     = (const float*)d_in[5];
    const float* w9         = (const float*)d_in[6];
    const float* bias       = (const float*)d_in[7];

    // d_out = [ pred f32 | list_feat 18x f32 | offset1 f16 | aff1 f16 ], flat
    float* outf = (float*)d_out;
    float* pred = outf;
    float* list = outf + (size_t)NPIX;
    f16*   off1 = (f16*)(outf + (size_t)19 * NPIX);
    f16*   aff1 = off1 + (size_t)18 * NPIX;

    dim3 cgrid(NTX, H_ / ROWS, B_);
    conv_offset_aff_wmma<<<cgrid, 128, 0, stream>>>(guidance, conv_w, conv_b, off1, aff1);

    dim3 pgrid(W_ / 64, H_, B_);
    for (int it = 0; it < 18; ++it) {
        const float* src = (it == 0) ? feat_init : (list + (size_t)(it - 1) * NPIX);
        float*       dst = list + (size_t)it * NPIX;
        prop_step<<<pgrid, 64, 0, stream>>>(src, confidence, feat_fix, off1, aff1,
                                            w9, bias, dst, (it == 17) ? pred : nullptr);
    }
}
